// TransformerBlock_69853348102788
// MI455X (gfx1250) — matrix operand order repaired and checked by reading, not compiled, not run on hardware
//
#include <hip/hip_runtime.h>
#include <hip/hip_bf16.h>

typedef __attribute__((ext_vector_type(16))) __bf16 v16bf;
typedef __attribute__((ext_vector_type(8)))  float  v8f;
typedef unsigned short ushort_t;
typedef int vint4 __attribute__((__vector_size__(16)));

#define EMB 1024
#define FFN_DIM 4096
#define NH 16
#define HD 64
#define SEQ 2048
#define BATCH 2
#define MTOT (BATCH*SEQ)   // 4096 tokens

#if defined(__AMDGCN__) && __has_builtin(__builtin_amdgcn_global_load_async_to_lds_b128) && __has_builtin(__builtin_amdgcn_s_wait_asynccnt)
#define USE_ASYNC_LDS 1
#define GPTR(p) ((vint4 __attribute__((address_space(1)))*)(p))
#define LPTR(p) ((vint4 __attribute__((address_space(3)))*)(p))
#else
#define USE_ASYNC_LDS 0
#endif

union FragBF {
    v16bf bf;
    uint4 q[2];
    ushort_t s[16];
};

__device__ __forceinline__ ushort_t f2bf(float f) {
    unsigned u = __float_as_uint(f);
    unsigned r = u + 0x7fffu + ((u >> 16) & 1u);   // round-to-nearest-even
    return (ushort_t)(r >> 16);
}

// ---------------------------------------------------------------- convert
__global__ __launch_bounds__(256) void k_f32_to_bf16(const float* __restrict__ s,
                                                     ushort_t* __restrict__ d, int n) {
    int i = blockIdx.x * 256 + threadIdx.x;
    if (i < n) d[i] = f2bf(s[i]);
}

// ---------------------------------------------------------------- layernorm -> bf16
__global__ __launch_bounds__(256) void k_layernorm_bf16(const float* __restrict__ x,
                                                        const float* __restrict__ g,
                                                        const float* __restrict__ b,
                                                        ushort_t* __restrict__ out) {
    __shared__ float red[256];
    const int row = blockIdx.x;            // token
    const int tid = threadIdx.x;
    const float* xr = x + (size_t)row * EMB;
    float4 xv = ((const float4*)xr)[tid];

    float s = xv.x + xv.y + xv.z + xv.w;
    red[tid] = s; __syncthreads();
    for (int off = 128; off > 0; off >>= 1) {
        if (tid < off) red[tid] += red[tid + off];
        __syncthreads();
    }
    const float mean = red[0] * (1.0f / EMB);
    __syncthreads();

    float dx = xv.x - mean, dy = xv.y - mean, dz = xv.z - mean, dw = xv.w - mean;
    red[tid] = dx*dx + dy*dy + dz*dz + dw*dw; __syncthreads();
    for (int off = 128; off > 0; off >>= 1) {
        if (tid < off) red[tid] += red[tid + off];
        __syncthreads();
    }
    const float rstd = rsqrtf(red[0] * (1.0f / EMB) + 1e-6f);

    float4 gv = ((const float4*)g)[tid];
    float4 bv = ((const float4*)b)[tid];
    ushort_t* o = out + (size_t)row * EMB + tid * 4;
    o[0] = f2bf(dx * rstd * gv.x + bv.x);
    o[1] = f2bf(dy * rstd * gv.y + bv.y);
    o[2] = f2bf(dz * rstd * gv.z + bv.z);
    o[3] = f2bf(dw * rstd * gv.w + bv.w);
}

// ---------------------------------------------------------------- bf16 GEMM  C = A * W^T (+epilogue)
// A: [M,K] bf16 row-major.  W: [N,K] bf16 row-major (torch-style weight).
// MODE 0: out bf16 = acc + bias                       (Q, K projections)
// MODE 1: out bf16 = leaky(acc + bias)                (FFN1)
// MODE 2: out f32  = acc + bias + resid               (Wo, FFN2)
// MODE 3: out bf16 = acc + bias, scattered to V^T layout [(b*NH+h)*HD+d][SEQ] (V projection)
template <int MODE>
__global__ __launch_bounds__(256) void k_gemm_bf16(const ushort_t* __restrict__ A,
                                                   const ushort_t* __restrict__ W,
                                                   const float* __restrict__ bias,
                                                   const float* __restrict__ resid,
                                                   void* __restrict__ outp,
                                                   int M, int N, int K) {
    constexpr int LDT = 40;                 // padded row stride (bf16 elems), 80B
    __shared__ ushort_t As[2][128 * LDT];   // double-buffered tiles
    __shared__ ushort_t Bs[2][128 * LDT];

    const int tid  = threadIdx.x;
    const int lane = tid & 31;
    const int wave = tid >> 5;
    const int wm   = wave >> 1;             // 0..3  -> 32 rows each
    const int wn   = wave & 1;              // 0..1  -> 64 cols each
    const int m0   = blockIdx.y * 128;
    const int n0   = blockIdx.x * 128;

    v8f acc[2][4];
#pragma unroll
    for (int i = 0; i < 2; ++i)
#pragma unroll
        for (int j = 0; j < 4; ++j) acc[i][j] = (v8f)0.0f;

    const int ldr = tid >> 1;               // 0..127
    const int ldc = (tid & 1) * 16;         // 0 or 16

    // Issue the cooperative copy of one 128x32 K-slice into LDS buffer `buf`.
    auto issue_copy = [&](int buf, int k0) {
        const ushort_t* ga = A + (size_t)(m0 + ldr) * K + k0 + ldc;
        const ushort_t* gb = W + (size_t)(n0 + ldr) * K + k0 + ldc;
        ushort_t* la = &As[buf][ldr * LDT + ldc];
        ushort_t* lb = &Bs[buf][ldr * LDT + ldc];
#if USE_ASYNC_LDS
        // memory -> LDS DMA, no VGPR round-trip; tracked by ASYNCcnt
        __builtin_amdgcn_global_load_async_to_lds_b128(GPTR(ga),     LPTR(la),     0, 0);
        __builtin_amdgcn_global_load_async_to_lds_b128(GPTR(ga + 8), LPTR(la + 8), 0, 0);
        __builtin_amdgcn_global_load_async_to_lds_b128(GPTR(gb),     LPTR(lb),     0, 0);
        __builtin_amdgcn_global_load_async_to_lds_b128(GPTR(gb + 8), LPTR(lb + 8), 0, 0);
#else
        *(uint4*)la       = *(const uint4*)ga;
        *(uint4*)(la + 8) = *(const uint4*)(ga + 8);
        *(uint4*)lb       = *(const uint4*)gb;
        *(uint4*)(lb + 8) = *(const uint4*)(gb + 8);
#endif
    };

    issue_copy(0, 0);
#if USE_ASYNC_LDS
    __builtin_amdgcn_s_wait_asynccnt(0);
#endif
    __syncthreads();

    int sbuf = 0;
    for (int k0 = 0; k0 < K; k0 += 32) {
        if (k0 + 32 < K) issue_copy(sbuf ^ 1, k0 + 32);   // prefetch next slice

        FragBF af[2], bfr[4];
        const int abase = (lane < 16) ? 0 : 8;     // A: k-chunk per half-wave
        const int bbase = (lane < 16) ? 0 : 8;     // B: k-chunk per half-wave
#pragma unroll
        for (int i = 0; i < 2; ++i) {
            const ushort_t* pa = &As[sbuf][(wm * 32 + i * 16 + (lane & 15)) * LDT + abase];
            af[i].q[0] = *(const uint4*)pa;
            af[i].q[1] = *(const uint4*)(pa + 16);
        }
#pragma unroll
        for (int j = 0; j < 4; ++j) {
            const ushort_t* pb = &Bs[sbuf][(wn * 64 + j * 16 + (lane & 15)) * LDT + bbase];
            bfr[j].q[0] = *(const uint4*)pb;
            bfr[j].q[1] = *(const uint4*)(pb + 16);
        }
#pragma unroll
        for (int i = 0; i < 2; ++i)
#pragma unroll
            for (int j = 0; j < 4; ++j)
                acc[i][j] = __builtin_amdgcn_wmma_f32_16x16x32_bf16(
                    false, af[i].bf, false, bfr[j].bf, (short)0, acc[i][j], false, false);

#if USE_ASYNC_LDS
        __builtin_amdgcn_s_wait_asynccnt(0);
#endif
        __syncthreads();
        sbuf ^= 1;
    }

    // epilogue: C/D layout — lane l, vgpr r -> (M = r + 8*(l>=16), N = l&15)
    const int nloc  = lane & 15;
    const int rbase = (lane < 16) ? 0 : 8;
#pragma unroll
    for (int i = 0; i < 2; ++i) {
#pragma unroll
        for (int j = 0; j < 4; ++j) {
            const int browb = m0 + wm * 32 + i * 16 + rbase;
            const int col   = n0 + wn * 64 + j * 16 + nloc;
#pragma unroll
            for (int r = 0; r < 8; ++r) {
                const int row = browb + r;
                float v = acc[i][j][r] + bias[col];
                if (MODE == 0) {
                    ((ushort_t*)outp)[(size_t)row * N + col] = f2bf(v);
                } else if (MODE == 1) {
                    v = (v >= 0.0f) ? v : 0.01f * v;
                    ((ushort_t*)outp)[(size_t)row * N + col] = f2bf(v);
                } else if (MODE == 2) {
                    ((float*)outp)[(size_t)row * N + col] = v + resid[(size_t)row * N + col];
                } else { // MODE 3: V^T head layout
                    const int bb  = row >> 11;       // batch
                    const int tok = row & (SEQ - 1);
                    const int h   = col >> 6;
                    const int d   = col & (HD - 1);
                    ((ushort_t*)outp)[((size_t)((bb * NH + h) * HD + d)) * SEQ + tok] = f2bf(v);
                }
            }
        }
    }
}

// ---------------------------------------------------------------- attention
// ctx[i, h*64+d] = sum_j leaky( (q_i . k_j) / 8 ) * v[j,d]   (per batch/head)
// q,k: flat [MTOT, EMB] bf16.  vT: [(b*NH+h)*HD+d][SEQ] bf16.  ctx out: flat bf16.
__global__ __launch_bounds__(256) void k_attention(const ushort_t* __restrict__ qf,
                                                   const ushort_t* __restrict__ kf,
                                                   const ushort_t* __restrict__ vT,
                                                   ushort_t* __restrict__ ctx) {
    __shared__ ushort_t probs[8][16 * 40];   // per-wave 16x32 bf16 probs tile (pad 8)

    const int tid  = threadIdx.x;
    const int lane = tid & 31;
    const int wave = tid >> 5;
    const int bh   = blockIdx.y;             // 0..31
    const int bb   = bh >> 4;
    const int h    = bh & 15;
    const int i0   = blockIdx.x * 128 + wave * 16;   // row-tile base within sequence

    // q fragments (A layout), K = HD = 64 -> two 16x32 frags
    FragBF aq[2];
    {
        const ushort_t* qrow = qf + (size_t)(bb * SEQ + i0 + (lane & 15)) * EMB + h * HD;
        const int half = (lane < 16) ? 0 : 8;
#pragma unroll
        for (int kk = 0; kk < 2; ++kk) {
            const ushort_t* p = qrow + kk * 32 + half;
            aq[kk].q[0] = *(const uint4*)p;
            aq[kk].q[1] = *(const uint4*)(p + 16);
        }
    }

    v8f cacc[4];
#pragma unroll
    for (int d = 0; d < 4; ++d) cacc[d] = (v8f)0.0f;

    const int nloc  = lane & 15;
    const int rbase = (lane < 16) ? 0 : 8;
    const int khalf = (lane < 16) ? 0 : 8;

    for (int j0 = 0; j0 < SEQ; j0 += 32) {
        // ---- scores for 16 x 32 window, two 16-column halves
#pragma unroll
        for (int jj = 0; jj < 2; ++jj) {
            v8f s = (v8f)0.0f;
#pragma unroll
            for (int kk = 0; kk < 2; ++kk) {
                FragBF bk;
                const ushort_t* p = kf + (size_t)(bb * SEQ + j0 + jj * 16 + nloc) * EMB
                                       + h * HD + kk * 32 + khalf;
                bk.q[0] = *(const uint4*)p;
                bk.q[1] = *(const uint4*)(p + 16);
                s = __builtin_amdgcn_wmma_f32_16x16x32_bf16(
                        false, aq[kk].bf, false, bk.bf, (short)0, s, false, false);
            }
            // scale + leaky_relu, C-layout -> probs LDS tile
#pragma unroll
            for (int r = 0; r < 8; ++r) {
                float v = s[r] * 0.125f;
                v = (v >= 0.0f) ? v : 0.01f * v;
                probs[wave][(rbase + r) * 40 + jj * 16 + nloc] = f2bf(v);
            }
        }

        // ---- reload probs as A fragment (16x32)
        FragBF ap;
        {
            const ushort_t* p = &probs[wave][(lane & 15) * 40 + ((lane < 16) ? 0 : 8)];
            ap.q[0] = *(const uint4*)p;
            ap.q[1] = *(const uint4*)(p + 16);
        }

        // ---- ctx += probs * V   (B fragments from vT rows: contiguous over j)
#pragma unroll
        for (int dd = 0; dd < 4; ++dd) {
            FragBF bv;
            const ushort_t* p = vT + ((size_t)((bb * NH + h) * HD + dd * 16 + nloc)) * SEQ
                                   + j0 + khalf;
            bv.q[0] = *(const uint4*)p;
            bv.q[1] = *(const uint4*)(p + 16);
            cacc[dd] = __builtin_amdgcn_wmma_f32_16x16x32_bf16(
                           false, ap.bf, false, bv.bf, (short)0, cacc[dd], false, false);
        }
    }

    // ---- writeback ctx (flat bf16 [MTOT, EMB])
#pragma unroll
    for (int dd = 0; dd < 4; ++dd) {
#pragma unroll
        for (int r = 0; r < 8; ++r) {
            const int tok = bb * SEQ + i0 + rbase + r;
            const int col = h * HD + dd * 16 + nloc;
            ctx[(size_t)tok * EMB + col] = f2bf(cacc[dd][r]);
        }
    }
}

// ---------------------------------------------------------------- launch
extern "C" void kernel_launch(void* const* d_in, const int* in_sizes, int n_in,
                              void* d_out, int out_size, void* d_ws, size_t ws_size,
                              hipStream_t stream) {
    const float* x     = (const float*)d_in[0];
    const float* ln1_g = (const float*)d_in[1];
    const float* ln1_b = (const float*)d_in[2];
    const float* ln2_g = (const float*)d_in[3];
    const float* ln2_b = (const float*)d_in[4];
    const float* Wq = (const float*)d_in[5];  const float* bq = (const float*)d_in[6];
    const float* Wk = (const float*)d_in[7];  const float* bk = (const float*)d_in[8];
    const float* Wv = (const float*)d_in[9];  const float* bv = (const float*)d_in[10];
    const float* Wo = (const float*)d_in[11]; const float* bo = (const float*)d_in[12];
    const float* W1 = (const float*)d_in[13]; const float* b1 = (const float*)d_in[14];
    const float* W2 = (const float*)d_in[15]; const float* b2 = (const float*)d_in[16];
    float* out = (float*)d_out;

    char* ws = (char*)d_ws;
    const size_t MB = 1024 * 1024;
    ushort_t* Wq_bf = (ushort_t*)(ws + 0 * MB);    // 2 MB
    ushort_t* Wk_bf = (ushort_t*)(ws + 2 * MB);    // 2 MB
    ushort_t* Wv_bf = (ushort_t*)(ws + 4 * MB);    // 2 MB
    ushort_t* Wo_bf = (ushort_t*)(ws + 6 * MB);    // 2 MB
    ushort_t* W1_bf = (ushort_t*)(ws + 8 * MB);    // 8 MB
    ushort_t* W2_bf = (ushort_t*)(ws + 16 * MB);   // 8 MB
    ushort_t* xn    = (ushort_t*)(ws + 24 * MB);   // 8 MB (reused for LN1 & LN2)
    ushort_t* qf    = (ushort_t*)(ws + 32 * MB);   // 8 MB
    ushort_t* kfb   = (ushort_t*)(ws + 40 * MB);   // 8 MB
    ushort_t* vT    = (ushort_t*)(ws + 48 * MB);   // 8 MB
    ushort_t* ctx   = (ushort_t*)(ws + 56 * MB);   // 8 MB
    float*    x2    = (float*)   (ws + 64 * MB);   // 16 MB
    ushort_t* f1    = (ushort_t*)(ws + 80 * MB);   // 32 MB

    const int nW  = EMB * EMB;         // 1M
    const int nW1 = FFN_DIM * EMB;     // 4M

    // 1) weights -> bf16
    k_f32_to_bf16<<<(nW  + 255) / 256, 256, 0, stream>>>(Wq, Wq_bf, nW);
    k_f32_to_bf16<<<(nW  + 255) / 256, 256, 0, stream>>>(Wk, Wk_bf, nW);
    k_f32_to_bf16<<<(nW  + 255) / 256, 256, 0, stream>>>(Wv, Wv_bf, nW);
    k_f32_to_bf16<<<(nW  + 255) / 256, 256, 0, stream>>>(Wo, Wo_bf, nW);
    k_f32_to_bf16<<<(nW1 + 255) / 256, 256, 0, stream>>>(W1, W1_bf, nW1);
    k_f32_to_bf16<<<(nW1 + 255) / 256, 256, 0, stream>>>(W2, W2_bf, nW1);

    // 2) LN1
    k_layernorm_bf16<<<MTOT, 256, 0, stream>>>(x, ln1_g, ln1_b, xn);

    // 3) Q, K, V projections
    dim3 gProj(EMB / 128, MTOT / 128);
    k_gemm_bf16<0><<<gProj, 256, 0, stream>>>(xn, Wq_bf, bq, nullptr, qf,  MTOT, EMB, EMB);
    k_gemm_bf16<0><<<gProj, 256, 0, stream>>>(xn, Wk_bf, bk, nullptr, kfb, MTOT, EMB, EMB);
    k_gemm_bf16<3><<<gProj, 256, 0, stream>>>(xn, Wv_bf, bv, nullptr, vT,  MTOT, EMB, EMB);

    // 4) attention
    k_attention<<<dim3(SEQ / 128, BATCH * NH), 256, 0, stream>>>(qf, kfb, vT, ctx);

    // 5) output projection + residual
    k_gemm_bf16<2><<<gProj, 256, 0, stream>>>(ctx, Wo_bf, bo, x, x2, MTOT, EMB, EMB);

    // 6) LN2
    k_layernorm_bf16<<<MTOT, 256, 0, stream>>>(x2, ln2_g, ln2_b, xn);

    // 7) FFN
    dim3 gF1(FFN_DIM / 128, MTOT / 128);
    k_gemm_bf16<1><<<gF1, 256, 0, stream>>>(xn, W1_bf, b1, nullptr, f1, MTOT, FFN_DIM, EMB);
    k_gemm_bf16<2><<<gProj, 256, 0, stream>>>(f1, W2_bf, b2, x2, out, MTOT, EMB, FFN_DIM);
}
